// FusionDecoder_53077205844793
// MI455X (gfx1250) — compile-verified, matrix operand order checked by reading, not run on hardware
//
#include <hip/hip_runtime.h>
#include <hip/hip_bf16.h>

typedef __attribute__((ext_vector_type(16))) _Float16 v16h;
typedef __attribute__((ext_vector_type(8)))  _Float16 v8h;
typedef __attribute__((ext_vector_type(8)))  float    v8f;

#define BN_EPS 1e-5f
#define DIM    64
#define CTX    32
#define K1     288   // 4*DIM + CTX
#define H1     128
#define H2     64
#define WAVES  4
#define EPW    16                 // edges per wave
#define EPB    (WAVES * EPW)      // 64 edges per block

// ---------------- workspace layout (bytes) ----------------
// [0, 73728)          W1h : f16 [H1][K1]  (transposed, BN-scale folded)
// [73728, 90112)      W2h : f16 [H2][H1]  (transposed, BN-scale folded)
// [90112, 90624)      t1  : f32 [H1]
// [90624, 90880)      t2  : f32 [H2]

__global__ void prep_weights_kernel(const float* __restrict__ W1, const float* __restrict__ b1,
                                    const float* __restrict__ g1, const float* __restrict__ be1,
                                    const float* __restrict__ m1, const float* __restrict__ v1,
                                    const float* __restrict__ W2, const float* __restrict__ b2,
                                    const float* __restrict__ g2, const float* __restrict__ be2,
                                    const float* __restrict__ m2, const float* __restrict__ v2,
                                    _Float16* __restrict__ W1h, _Float16* __restrict__ W2h,
                                    float* __restrict__ t1, float* __restrict__ t2) {
  int tid = blockIdx.x * blockDim.x + threadIdx.x;
  if (tid < H1 * K1) {
    int n = tid / K1, k = tid % K1;
    float s = g1[n] * rsqrtf(v1[n] + BN_EPS);
    W1h[tid] = (_Float16)(W1[k * H1 + n] * s);
  } else if (tid < H1 * K1 + H2 * H1) {
    int i = tid - H1 * K1;
    int n = i / H1, k = i % H1;
    float s = g2[n] * rsqrtf(v2[n] + BN_EPS);
    W2h[i] = (_Float16)(W2[k * H2 + n] * s);
  } else if (tid < H1 * K1 + H2 * H1 + H1) {
    int n = tid - H1 * K1 - H2 * H1;
    float s = g1[n] * rsqrtf(v1[n] + BN_EPS);
    t1[n] = (b1[n] - m1[n]) * s + be1[n];
  } else if (tid < H1 * K1 + H2 * H1 + H1 + H2) {
    int n = tid - H1 * K1 - H2 * H1 - H1;
    float s = g2[n] * rsqrtf(v2[n] + BN_EPS);
    t2[n] = (b2[n] - m2[n]) * s + be2[n];
  }
}

// Load a 16-bit A/B WMMA fragment for this lane from a row-major matrix.
// rowp = pointer to this lane's row (row = lane%16); k0 = kchunk*32 + (lane/16)*8.
// VGPRs 0..3 <- K = k0..k0+7 ; VGPRs 4..7 <- K = k0+16..k0+23  (two b128 loads)
__device__ __forceinline__ v16h load_frag(const _Float16* rowp, int k0) {
  v8h lo = *(const v8h*)(rowp + k0);
  v8h up = *(const v8h*)(rowp + k0 + 16);
  v16h r;
#pragma unroll
  for (int i = 0; i < 8; ++i) { r[i] = lo[i]; r[8 + i] = up[i]; }
  return r;
}

__global__ __launch_bounds__(WAVES * 32)
void edge_mlp_wmma_kernel(const float* __restrict__ z,
                          const long long* __restrict__ eidx,   // [2][E] int64
                          const float* __restrict__ ectx,       // [E][CTX]
                          const _Float16* __restrict__ W1h,
                          const _Float16* __restrict__ W2h,
                          const float* __restrict__ t1,
                          const float* __restrict__ t2,
                          const float* __restrict__ W3,         // [H2]
                          const float* __restrict__ b3,
                          float* __restrict__ out,
                          long long E) {
  __shared__ __attribute__((aligned(16))) _Float16 shFused[WAVES][EPW][K1]; // 36864 B
  __shared__ __attribute__((aligned(16))) _Float16 shX1[WAVES][EPW][H1];    // 16384 B
  __shared__ __attribute__((aligned(16))) _Float16 shX2[WAVES][EPW][H2];    //  8192 B

  const int lane = threadIdx.x & 31;
  const int wave = threadIdx.x >> 5;
  const long long ebase = (long long)blockIdx.x * EPB + (long long)wave * EPW;

  _Float16 (*fused)[K1] = shFused[wave];
  _Float16 (*x1)[H1]    = shX1[wave];
  _Float16 (*x2)[H2]    = shX2[wave];

  // ---- Phase 0: gather endpoints, build fused features (2 lanes per edge) ----
  {
    const int m    = lane >> 1;
    const int part = lane & 1;
    long long e  = ebase + m;
    long long ec = (e < E) ? e : (E - 1);   // clamp; store is guarded later
    long long i0 = eidx[ec];
    long long i1 = eidx[E + ec];
    const float* zs = z + i0 * DIM + part * 32;
    const float* zd = z + i1 * DIM + part * 32;
    _Float16* fr = fused[m];
#pragma unroll
    for (int k = 0; k < 32; ++k) {
      float s = zs[k], d = zd[k];
      fr[part * 32 + k]       = (_Float16)s;
      fr[64  + part * 32 + k] = (_Float16)d;
      fr[128 + part * 32 + k] = (_Float16)(s * d);
      fr[192 + part * 32 + k] = (_Float16)fabsf(s - d);
    }
    const float* ep = ectx + ec * CTX + part * 16;
#pragma unroll
    for (int k = 0; k < 16; ++k)
      fr[256 + part * 16 + k] = (_Float16)ep[k];
  }
  __syncthreads();

  const int row    = lane & 15;        // A: M row / B: N column for this lane
  const int khalf  = (lane >> 4) * 8;  // K offset within chunk per lane half
  const int mhalf  = lane >> 4;        // C/D: rows r+8*mhalf

  // ---- Layer 1: [16 x 288] x [288 x 128] via 9 K-chunks x 8 N-tiles ----
  v8f acc1[8];
#pragma unroll
  for (int nt = 0; nt < 8; ++nt) {
    v8f zf = {0.f, 0.f, 0.f, 0.f, 0.f, 0.f, 0.f, 0.f};
    acc1[nt] = zf;
  }
  const _Float16* aRow1 = fused[row];
#pragma unroll
  for (int kc = 0; kc < K1 / 32; ++kc) {
    const int k0 = kc * 32 + khalf;
    v16h a = load_frag(aRow1, k0);
#pragma unroll
    for (int nt = 0; nt < 8; ++nt) {
      v16h b = load_frag(W1h + (nt * 16 + row) * K1, k0);
      acc1[nt] = __builtin_amdgcn_wmma_f32_16x16x32_f16(
          false, a, false, b, (short)0, acc1[nt], false, false);
    }
  }
  // bias + ReLU -> x1 (f16, row-major) ; D layout: VGPR r = row r+8*mhalf, lane%16 = col
#pragma unroll
  for (int nt = 0; nt < 8; ++nt) {
    const int n = nt * 16 + row;
    const float t = t1[n];
#pragma unroll
    for (int r = 0; r < 8; ++r) {
      float v = acc1[nt][r] + t;
      v = v > 0.f ? v : 0.f;
      x1[r + 8 * mhalf][n] = (_Float16)v;
    }
  }
  __syncthreads();

  // ---- Layer 2: [16 x 128] x [128 x 64] via 4 K-chunks x 4 N-tiles ----
  v8f acc2[4];
#pragma unroll
  for (int nt = 0; nt < 4; ++nt) {
    v8f zf = {0.f, 0.f, 0.f, 0.f, 0.f, 0.f, 0.f, 0.f};
    acc2[nt] = zf;
  }
  const _Float16* aRow2 = x1[row];
#pragma unroll
  for (int kc = 0; kc < H1 / 32; ++kc) {
    const int k0 = kc * 32 + khalf;
    v16h a = load_frag(aRow2, k0);
#pragma unroll
    for (int nt = 0; nt < 4; ++nt) {
      v16h b = load_frag(W2h + (nt * 16 + row) * H1, k0);
      acc2[nt] = __builtin_amdgcn_wmma_f32_16x16x32_f16(
          false, a, false, b, (short)0, acc2[nt], false, false);
    }
  }
#pragma unroll
  for (int nt = 0; nt < 4; ++nt) {
    const int n = nt * 16 + row;
    const float t = t2[n];
#pragma unroll
    for (int r = 0; r < 8; ++r) {
      float v = acc2[nt][r] + t;
      v = v > 0.f ? v : 0.f;
      x2[r + 8 * mhalf][n] = (_Float16)v;
    }
  }
  __syncthreads();

  // ---- Layer 3: [16 x 64] x [64 x 1] in fp32 (one edge per lane, lanes 0..15) ----
  if (lane < 16) {
    const long long e = ebase + lane;
    if (e < E) {
      float acc = b3[0];
      const _Float16* xr = x2[lane];
#pragma unroll
      for (int k = 0; k < H2; ++k) acc += (float)xr[k] * W3[k];
      out[e] = acc;
    }
  }
}

extern "C" void kernel_launch(void* const* d_in, const int* in_sizes, int n_in,
                              void* d_out, int out_size, void* d_ws, size_t ws_size,
                              hipStream_t stream) {
  const float*     z    = (const float*)d_in[0];
  const long long* eidx = (const long long*)d_in[1];   // int64 [2,E]
  const float*     ectx = (const float*)d_in[2];
  const float*     W1   = (const float*)d_in[3];
  const float*     b1   = (const float*)d_in[4];
  const float*     g1   = (const float*)d_in[5];
  const float*     be1  = (const float*)d_in[6];
  const float*     m1   = (const float*)d_in[7];
  const float*     v1   = (const float*)d_in[8];
  const float*     W2   = (const float*)d_in[9];
  const float*     b2   = (const float*)d_in[10];
  const float*     g2   = (const float*)d_in[11];
  const float*     be2  = (const float*)d_in[12];
  const float*     m2   = (const float*)d_in[13];
  const float*     v2   = (const float*)d_in[14];
  const float*     W3   = (const float*)d_in[15];
  const float*     b3   = (const float*)d_in[16];
  float* out = (float*)d_out;

  const long long E = (long long)in_sizes[1] / 2;

  _Float16* W1h = (_Float16*)d_ws;                         // 36864 halves
  _Float16* W2h = W1h + H1 * K1;                           //  8192 halves
  float*    t1  = (float*)((char*)d_ws + 90112);
  float*    t2  = t1 + H1;

  {
    const int total  = H1 * K1 + H2 * H1 + H1 + H2;        // 45248
    const int thr    = 256;
    const int blocks = (total + thr - 1) / thr;
    prep_weights_kernel<<<blocks, thr, 0, stream>>>(
        W1, b1, g1, be1, m1, v1, W2, b2, g2, be2, m2, v2, W1h, W2h, t1, t2);
  }
  {
    const long long blocks = (E + EPB - 1) / EPB;
    edge_mlp_wmma_kernel<<<(unsigned)blocks, WAVES * 32, 0, stream>>>(
        z, eidx, ectx, W1h, W2h, t1, t2, W3, b3, out, E);
  }
}
